// GaussianImage_83124797047011
// MI455X (gfx1250) — matrix operand order repaired and checked by reading, not compiled, not run on hardware
//
#include <hip/hip_runtime.h>

#define NG     512
#define PLANE  512
#define LOG2E  1.4426950408889634f
#define TWO_PI 6.283185307179586f

typedef __attribute__((ext_vector_type(2)))  float    v2f;
typedef __attribute__((ext_vector_type(8)))  float    v8f;
typedef __attribute__((ext_vector_type(16))) _Float16 v16h;
typedef __attribute__((ext_vector_type(4)))  unsigned v4u;

__device__ __forceinline__ float fexp2(float x) { return __builtin_amdgcn_exp2f(x); }

// monotone float <-> uint mapping so unsigned atomicMax == float max (handles negatives)
__device__ __forceinline__ unsigned enc_f32(float f) {
  unsigned u = __float_as_uint(f);
  return (u & 0x80000000u) ? ~u : (u | 0x80000000u);
}
__device__ __forceinline__ float dec_f32(unsigned k) {
  return (k & 0x80000000u) ? __uint_as_float(k ^ 0x80000000u) : __uint_as_float(~k);
}

// ---------------------------------------------------------------------------
// Prep: per-gaussian polynomial coeffs (log2 domain, -0.5*log2e folded in),
// laid out as WMMA A-fragments: a1 = f32 16x4 A layout (gaussians = M rows),
// a2 = f16 16x32 A layout of W^T (rgb*alpha), coef = plain copy for max pass.
// ---------------------------------------------------------------------------
__global__ __launch_bounds__(256) void gs_prep(
    const float* __restrict__ mean, const float* __restrict__ alpha,
    const float* __restrict__ scale, const float* __restrict__ theta,
    const float* __restrict__ rgb,
    float* __restrict__ a1, unsigned* __restrict__ a2,
    float* __restrict__ coef, unsigned* __restrict__ maxcell) {
  int t = blockIdx.x * 256 + threadIdx.x;       // 4096 threads
  if (t == 0) maxcell[0] = 0u;                  // < enc(-inf): init for atomicMax

  if (t < NG) {
    int n = t;
    float th = TWO_PI * theta[n];
    float cs = __cosf(th), sn = __sinf(th);
    float s0 = scale[2*n], s1 = scale[2*n+1];
    float s0q = s0*s0, s1q = s1*s1;
    float a  = cs*cs*s0q + sn*sn*s1q;
    float b  = cs*sn*(s0q - s1q);
    float dd = sn*sn*s0q + cs*cs*s1q;
    float det = a*dd - b*b;
    float i00 = dd/det, i11 = a/det;
    float Bq  = -2.0f*b/det;                    // inv01 + inv10
    float mux = mean[2*n], muy = mean[2*n+1];
    const float k = -0.5f * LOG2E;
    float A_ = k*i00;
    float B_ = k*Bq;
    float C_ = k*i11;
    float D_ = k*(-2.0f*i00*mux - Bq*muy);
    float E_ = k*(-Bq*mux - 2.0f*i11*muy);
    float log2c = -__log2f(TWO_PI) - 0.5f*__log2f(det);
    float F_ = k*(i00*mux*mux + Bq*mux*muy + i11*muy*muy) + log2c;
    float* cf = coef + n*8;
    cf[0]=A_; cf[1]=B_; cf[2]=C_; cf[3]=D_; cf[4]=E_; cf[5]=F_; cf[6]=0.f; cf[7]=0.f;
    // f32 16x4 A layout: lanes 0-15 get K={0,1} (wmma#1) / K={4,5} (wmma#2),
    // lanes 16-31 get K={2,3} / K={6,7}.  Features: x2,xy,y2,x,y,1,0,0
    int g = n >> 4, l = n & 15;
    float* lo = a1 + (g*32 + l)*4;
    float* hi = a1 + (g*32 + l + 16)*4;
    lo[0]=A_; lo[1]=B_; lo[2]=E_; lo[3]=F_;
    hi[0]=C_; hi[1]=D_; hi[2]=0.f; hi[3]=0.f;
  }

  // W^T fragments: chunk t>>8, lane (t>>3)&31, vgpr t&7 (f16 pairs)
  {
    int v = t & 7, L = (t >> 3) & 31, ch = t >> 8;
    int half = L >> 4;
    int Kb = (v < 4) ? (half*8 + 2*v) : (16 + half*8 + 2*(v-4));
    int n0 = ch*32 + Kb;
    int cc = L & 15;
    float w0 = (cc < 3) ? rgb[n0*3 + cc]     * alpha[n0]     : 0.0f;
    float w1 = (cc < 3) ? rgb[(n0+1)*3 + cc] * alpha[n0+1]   : 0.0f;
    auto h2 = __builtin_amdgcn_cvt_pkrtz(w0, w1);
    a2[ch*256 + L*8 + v] = __builtin_bit_cast(unsigned, h2);
  }
}

// ---------------------------------------------------------------------------
// Exact global max of log2(prob): per gaussian, sweep 512 rows; each row's 1-D
// slice is a concave parabola -> nearest clamped grid point to the vertex.
// ---------------------------------------------------------------------------
__global__ __launch_bounds__(256) void gs_maxlog(const float* __restrict__ coef,
                                                 unsigned* __restrict__ maxcell) {
  int n = blockIdx.x*256 + threadIdx.x;
  if (n >= NG) return;
  const float* c = coef + n*8;
  float A=c[0], B=c[1], C=c[2], D=c[3], E=c[4], F=c[5];   // A,C < 0
  float inv2A = -0.5f / A;
  float best = -3.0e38f;
  for (int j = 0; j < PLANE; ++j) {
    float y  = (float)j * (1.0f/511.0f);
    float by = B*y + D;
    float xs = by * inv2A;                                // vertex x
    float xg = fminf(fmaxf(rintf(xs * 511.0f), 0.0f), 511.0f) * (1.0f/511.0f);
    float lp = (A*xg + by)*xg + (C*y + E)*y + F;
    best = fmaxf(best, lp);
  }
  atomicMax(maxcell, enc_f32(best));
}

// ---------------------------------------------------------------------------
// Fixup: fold the global max into the constant-term coefficient (F, held in
// the low-lane A-fragment slot [3]) so the hot loop needs no per-element sub.
// a1 is rebuilt by gs_prep every launch, so this stays replay-deterministic.
// ---------------------------------------------------------------------------
__global__ __launch_bounds__(256) void gs_fixup(float* __restrict__ a1,
                                                const unsigned* __restrict__ maxcell) {
  int t = blockIdx.x*256 + threadIdx.x;         // 512 threads: 32 groups x 16 lanes
  if (t >= 512) return;
  int g = t >> 4, l = t & 15;
  float mx = dec_f32(maxcell[0]);
  a1[(g*32 + l)*4 + 3] -= mx;
}

// ---------------------------------------------------------------------------
// Main: one wave per 16-pixel row strip.
//   D1 (2x v_wmma_f32_16x16x4_f32) = log2prob - max, M=gaussian / N=pixel
//   splat = exp2(D1), pack f16, bounce through private LDS slice
//   acc (v_wmma_f32_16x16x32_f16) += W^T(16c x 32g) x splat(32g x 16px)
// ---------------------------------------------------------------------------
__global__ __launch_bounds__(128) void gs_splat(
    const float* __restrict__ a1, const unsigned* __restrict__ a2,
    float* __restrict__ out) {
  __shared__ unsigned lds[1024];                // 4 waves x 16px x 16 dwords
  int lane = threadIdx.x & 31;
  int wv   = threadIdx.x >> 5;
  int tile = blockIdx.x*4 + wv;                 // 16384 tiles = 512 rows x 32
  int ty = tile >> 5;
  int tx = tile & 31;
  int p   = lane & 15;
  int hiL = lane >> 4;
  float x = (float)(tx*16 + p) * (1.0f/511.0f);
  float y = (float)ty * (1.0f/511.0f);

  // f32 4x16 B layout (features x pixels): lanes 0-15 K={0,1}/{4,5}, 16-31 K={2,3}/{6,7}
  v2f b1a, b1b;
  b1a.x = hiL ? y*y : x*x;
  b1a.y = hiL ? x   : x*y;
  b1b.x = hiL ? 0.0f : y;
  b1b.y = hiL ? 0.0f : 1.0f;

  unsigned* row = lds + wv*256 + p*16;

  v8f acc = {0.f,0.f,0.f,0.f,0.f,0.f,0.f,0.f};

  for (int ch = 0; ch < 16; ++ch) {             // 32 gaussians per chunk
    #pragma unroll
    for (int g2 = 0; g2 < 2; ++g2) {
      int grp = ch*2 + g2;
      float4 f = *(const float4*)(a1 + (grp*32 + lane)*4);
      v2f a1a = {f.x, f.y};
      v2f a1b = {f.z, f.w};
      v8f d = {0.f,0.f,0.f,0.f,0.f,0.f,0.f,0.f};
      d = __builtin_amdgcn_wmma_f32_16x16x4_f32(false, a1a, false, b1a, (short)0, d, false, false);
      d = __builtin_amdgcn_wmma_f32_16x16x4_f32(false, a1b, false, b1b, (short)0, d, false, false);
      // d[r] = log2prob - max for gaussian grp*16 + 8*hiL + r at pixel p
      v4u pk;
      #pragma unroll
      for (int j = 0; j < 4; ++j) {
        float e0 = fexp2(d[2*j]);
        float e1 = fexp2(d[2*j+1]);
        auto h2 = __builtin_amdgcn_cvt_pkrtz(e0, e1);
        pk[j] = __builtin_bit_cast(unsigned, h2);
      }
      *(v4u*)(row + hiL*8 + g2*4) = pk;         // splat_lds[p][kk pairs], f16
    }
    // B fragment of splat (32g x 16px): lanes 0-15 K=0..7,16..23, lanes 16-31 K=8..15,24..31
    v16h b2  = *(const v16h*)(row + hiL*8);
    v16h a2v = *(const v16h*)(a2 + ch*256 + lane*8);
    acc = __builtin_amdgcn_wmma_f32_16x16x32_f16(false, a2v, false, b2, (short)0, acc, false, false);
  }

  // acc: M=channel (rows r + 8*hiL), N=pixel; only c=0..2 on lanes 0-15 useful
  if (lane < 16) {
    int base = (ty*PLANE + tx*16 + p) * 3;
    #pragma unroll
    for (int r = 0; r < 3; ++r) {
      float v = acc[r];
      out[base + r] = __builtin_amdgcn_rcpf(1.0f + fexp2(-LOG2E * v));
    }
  }
}

extern "C" void kernel_launch(void* const* d_in, const int* in_sizes, int n_in,
                              void* d_out, int out_size, void* d_ws, size_t ws_size,
                              hipStream_t stream) {
  const float* mean  = (const float*)d_in[0];
  const float* alpha = (const float*)d_in[1];
  const float* scale = (const float*)d_in[2];
  const float* theta = (const float*)d_in[3];
  const float* rgb   = (const float*)d_in[4];
  // d_in[5] (pixels) is an exact linspace grid; recomputed on device.
  float* out = (float*)d_out;

  char* ws = (char*)d_ws;                 // needs ~48 KB + 4 B of scratch
  float*    a1      = (float*)(ws + 0);          // 32 grp x 32 lane x 4 f32 = 16 KB
  unsigned* a2      = (unsigned*)(ws + 16384);   // 16 ch x 32 lane x 8 u32 = 16 KB
  float*    coef    = (float*)(ws + 32768);      // 512 x 8 f32            = 16 KB
  unsigned* maxcell = (unsigned*)(ws + 49152);   // global-max cell

  gs_prep  <<<16,   256, 0, stream>>>(mean, alpha, scale, theta, rgb, a1, a2, coef, maxcell);
  gs_maxlog<<<2,    256, 0, stream>>>(coef, maxcell);
  gs_fixup <<<2,    256, 0, stream>>>(a1, maxcell);
  gs_splat <<<4096, 128, 0, stream>>>(a1, a2, out);
}
